// GAT_31877247271007
// MI455X (gfx1250) — matrix operand order repaired and checked by reading, not compiled, not run on hardware
//
#include <hip/hip_runtime.h>
#include <hip/hip_bf16.h>
#include <math.h>

#define D_H    256
#define D_OUTC 64
#define MT     4          // M-tiles (16 rows each) per wave in the GEMM

typedef __attribute__((ext_vector_type(16))) __bf16 v16bf;
typedef __attribute__((ext_vector_type(8)))  __bf16 v8bf;
typedef __attribute__((ext_vector_type(8)))  float  v8f;

union V16U { v16bf v; v8bf h[2]; };
union V8FU { v8f v; float f[8]; };

// ---------------- elementwise helpers ----------------

__global__ void k_convert_bf16(const float* __restrict__ in, __bf16* __restrict__ out, size_t n) {
    size_t i = (size_t)blockIdx.x * blockDim.x + threadIdx.x;
    if (i < n) out[i] = (__bf16)in[i];
}

// W: [K x Nc] row-major fp32 -> WT: [Nc x K] bf16 (transposed, contiguous-K rows)
__global__ void k_transpose_bf16(const float* __restrict__ W, __bf16* __restrict__ WT, int K, int Nc) {
    int i = blockIdx.x * blockDim.x + threadIdx.x;
    if (i >= K * Nc) return;
    int n = i / K;
    int k = i - n * K;
    WT[(size_t)n * K + k] = (__bf16)W[(size_t)k * Nc + n];
}

__global__ void k_fill_u32(unsigned* __restrict__ p, unsigned v, size_t n) {
    size_t i = (size_t)blockIdx.x * blockDim.x + threadIdx.x;
    if (i < n) p[i] = v;
}

// ---------------- WMMA bf16 GEMM ----------------
// C[N x Kout] = act(A[N x 256] @ W[256 x Kout] + bias (+ res)), W given transposed bf16.
// Each wave computes MT(=4) 16x16 tiles stacked in M, sharing one B fragment per k-step:
// per k-step: 2 b128 B-loads + MT*2 b128 A-loads feed MT v_wmma_f32_16x16x32_bf16.
__global__ void k_gemm_bf16(const __bf16* __restrict__ A, const __bf16* __restrict__ WT,
                            const float* __restrict__ bias, const float* __restrict__ res,
                            float* __restrict__ Cf, __bf16* __restrict__ Cb,
                            int N, int Kout, int do_tanh) {
    const int lane = threadIdx.x & 31;
    const int wv   = threadIdx.x >> 5;
    const int wave = blockIdx.x * (blockDim.x >> 5) + wv;
    const int tn_cnt = Kout >> 4;
    const int tg_cnt = (N + (16 * MT) - 1) / (16 * MT);
    if (wave >= tg_cnt * tn_cnt) return;          // wave-uniform exit: EXEC stays all-1s
    const int tg = wave / tn_cnt;
    const int tn = wave - tg * tn_cnt;

    const int lm = lane & 15;
    const int h  = lane >> 4;
    const int row_base = tg * 16 * MT;

    const int col = tn * 16 + lm;
    const __bf16* brow = WT + (size_t)col * D_H;

    const __bf16* arow[MT];
#pragma unroll
    for (int mt = 0; mt < MT; ++mt) {
        int r = row_base + mt * 16 + lm;
        if (r > N - 1) r = N - 1;                 // clamp loads for the ragged tail
        arow[mt] = A + (size_t)r * D_H;
    }

    V8FU acc[MT];
#pragma unroll
    for (int mt = 0; mt < MT; ++mt)
#pragma unroll
        for (int r = 0; r < 8; ++r) acc[mt].f[r] = 0.0f;

#pragma unroll
    for (int kk = 0; kk < D_H; kk += 32) {
        V16U b;
        // B 32x16 frag: lane (col) holds K in [kk+16h .. kk+16h+15] (weights pre-transposed)
        b.h[0] = *(const v8bf*)(brow + kk + 16 * h);
        b.h[1] = *(const v8bf*)(brow + kk + 16 * h + 8);
#pragma unroll
        for (int mt = 0; mt < MT; ++mt) {
            V16U a;
            // A 16x32 frag: lane holds K in [kk+16h .. kk+16h+15]
            a.h[0] = *(const v8bf*)(arow[mt] + kk + 16 * h);
            a.h[1] = *(const v8bf*)(arow[mt] + kk + 16 * h + 8);
            acc[mt].v = __builtin_amdgcn_wmma_f32_16x16x32_bf16(
                            false, a.v, false, b.v, (short)0, acc[mt].v, false, false);
        }
    }

    const float bval = bias ? bias[col] : 0.0f;
#pragma unroll
    for (int mt = 0; mt < MT; ++mt) {
        const int row0 = row_base + mt * 16;
#pragma unroll
        for (int r = 0; r < 8; ++r) {
            const int row = row0 + r + 8 * h;     // C/D layout: VGPR r -> row r + 8*(lane>>4)
            if (row < N) {
                float x = acc[mt].f[r] + bval;
                if (res) x += res[(size_t)row * Kout + col];
                if (do_tanh) x = tanhf(x);
                Cf[(size_t)row * Kout + col] = x;
                if (Cb) Cb[(size_t)row * Kout + col] = (__bf16)x;
            }
        }
    }
}

// ---------------- edge kernels ----------------

__device__ __forceinline__ unsigned f32_order_encode(float f) {
    unsigned b = __float_as_uint(f);
    return b ^ ((b & 0x80000000u) ? 0xFFFFFFFFu : 0x80000000u);
}
__device__ __forceinline__ float f32_order_decode(unsigned k) {
    unsigned b = (k & 0x80000000u) ? (k ^ 0x80000000u) : ~k;
    return __uint_as_float(b);
}

// one wave per edge, lane owns channels [8*lane, 8*lane+8):
// t[e] = dot(v, tanh(src_proj[src] + dst_proj[dst])); segment max via uint atomicMax
__global__ void k_edge_scores(const float* __restrict__ sp, const float* __restrict__ dp,
                              const float* __restrict__ vv,
                              const int* __restrict__ es, const int* __restrict__ ed,
                              float* __restrict__ t, unsigned* __restrict__ segmax, int E) {
    int gw   = (int)(((size_t)blockIdx.x * blockDim.x + threadIdx.x) >> 5);
    int lane = threadIdx.x & 31;
    if (gw >= E) return;
    const int s = es[gw], d = ed[gw];
    const int base = lane * 8;                    // 32 lanes x 8 = 256 channels
    const float4* s4 = (const float4*)(sp + (size_t)s * D_H + base);
    const float4* d4 = (const float4*)(dp + (size_t)d * D_H + base);
    const float4* v4 = (const float4*)(vv + base);
    float sum = 0.0f;
#pragma unroll
    for (int q = 0; q < 2; ++q) {
        float4 a = s4[q], b = d4[q], w = v4[q];
        sum += tanhf(a.x + b.x) * w.x;
        sum += tanhf(a.y + b.y) * w.y;
        sum += tanhf(a.z + b.z) * w.z;
        sum += tanhf(a.w + b.w) * w.w;
    }
#pragma unroll
    for (int off = 16; off > 0; off >>= 1) sum += __shfl_xor(sum, off, 32);
    if (lane == 0) {
        t[gw] = sum;
        atomicMax(&segmax[d], f32_order_encode(sum));
    }
}

// t[e] <- exp(t[e] - max[dst]); denom[dst] += t[e]
__global__ void k_edge_exp(float* __restrict__ t, const int* __restrict__ ed,
                           const unsigned* __restrict__ segmax, float* __restrict__ denom, int E) {
    int e = blockIdx.x * blockDim.x + threadIdx.x;
    if (e >= E) return;
    const int d = ed[e];
    float ex = expf(t[e] - f32_order_decode(segmax[d]));
    t[e] = ex;
    atomicAdd(&denom[d], ex);
}

// one wave per edge, lane owns channels [8*lane, 8*lane+8):
// pool[dst] += (t[e]/denom[dst]) * cur[src]
__global__ void k_edge_pool(const float* __restrict__ t, const float* __restrict__ denom,
                            const int* __restrict__ es, const int* __restrict__ ed,
                            const float* __restrict__ cur, float* __restrict__ pool, int E) {
    int gw   = (int)(((size_t)blockIdx.x * blockDim.x + threadIdx.x) >> 5);
    int lane = threadIdx.x & 31;
    if (gw >= E) return;
    const int s = es[gw], d = ed[gw];
    const float a = t[gw] / denom[d];
    const int base = lane * 8;
    const float4* c4 = (const float4*)(cur + (size_t)s * D_H + base);
    float* pr = pool + (size_t)d * D_H + base;
    float4 x0 = c4[0], x1 = c4[1];
    atomicAdd(pr + 0, a * x0.x);
    atomicAdd(pr + 1, a * x0.y);
    atomicAdd(pr + 2, a * x0.z);
    atomicAdd(pr + 3, a * x0.w);
    atomicAdd(pr + 4, a * x1.x);
    atomicAdd(pr + 5, a * x1.y);
    atomicAdd(pr + 6, a * x1.z);
    atomicAdd(pr + 7, a * x1.w);
}

__global__ void k_select(const float* __restrict__ logits, const int* __restrict__ idx,
                         float* __restrict__ out, int nsel) {
    int g = blockIdx.x * blockDim.x + threadIdx.x;
    if (g >= nsel * D_OUTC) return;
    int i = g / D_OUTC, j = g - i * D_OUTC;
    out[g] = logits[(size_t)idx[i] * D_OUTC + j];
}

// ---------------- host orchestration ----------------

extern "C" void kernel_launch(void* const* d_in, const int* in_sizes, int n_in,
                              void* d_out, int out_size, void* d_ws, size_t ws_size,
                              hipStream_t stream) {
    (void)n_in; (void)out_size; (void)ws_size;
    const float* node_feat = (const float*)d_in[0];
    const float* W_x = (const float*)d_in[1];
    const float* b_x = (const float*)d_in[2];
    const float* Ws0 = (const float*)d_in[3];
    const float* Wd0 = (const float*)d_in[4];
    const float* v0  = (const float*)d_in[5];
    const float* Wn0 = (const float*)d_in[6];
    const float* bn0 = (const float*)d_in[7];
    const float* Ws1 = (const float*)d_in[8];
    const float* Wd1 = (const float*)d_in[9];
    const float* v1  = (const float*)d_in[10];
    const float* Wn1 = (const float*)d_in[11];
    const float* bn1 = (const float*)d_in[12];
    const float* Wo  = (const float*)d_in[13];
    const float* bo  = (const float*)d_in[14];
    const int* edge_src   = (const int*)d_in[15];
    const int* edge_dst   = (const int*)d_in[16];
    const int* select_idx = (const int*)d_in[17];

    const int N    = in_sizes[0] / D_H;
    const int E    = in_sizes[15];
    const int NSEL = in_sizes[17];
    float* out = (float*)d_out;

    // bump allocator over workspace
    char* ws = (char*)d_ws;
    size_t off = 0;
    auto alloc = [&](size_t bytes) -> void* {
        void* p = ws + off;
        off += (bytes + 255) & ~(size_t)255;
        return p;
    };

    __bf16* wtx  = (__bf16*)alloc((size_t)D_H * D_H * 2);
    __bf16* wts0 = (__bf16*)alloc((size_t)D_H * D_H * 2);
    __bf16* wtd0 = (__bf16*)alloc((size_t)D_H * D_H * 2);
    __bf16* wtn0 = (__bf16*)alloc((size_t)D_H * D_H * 2);
    __bf16* wts1 = (__bf16*)alloc((size_t)D_H * D_H * 2);
    __bf16* wtd1 = (__bf16*)alloc((size_t)D_H * D_H * 2);
    __bf16* wtn1 = (__bf16*)alloc((size_t)D_H * D_H * 2);
    __bf16* wto  = (__bf16*)alloc((size_t)D_H * D_OUTC * 2);
    __bf16* feat_bf = (__bf16*)alloc((size_t)N * D_H * 2);   // reused as pool_bf16 later
    float*  ne      = (float*) alloc((size_t)N * D_H * 4);   // node_embed (residual), cur for layer 0
    float*  curf    = (float*) alloc((size_t)N * D_H * 4);   // cur after layer updates
    __bf16* curb    = (__bf16*)alloc((size_t)N * D_H * 2);   // bf16 copy of cur (GEMM A operand)
    float*  sp      = (float*) alloc((size_t)N * D_H * 4);   // src_proj; reused as logits
    float*  dp      = (float*) alloc((size_t)N * D_H * 4);   // dst_proj; reused as pool
    float*  t       = (float*) alloc((size_t)E * 4);
    unsigned* segmax= (unsigned*)alloc((size_t)N * 4);
    float*  denom   = (float*) alloc((size_t)N * 4);

    auto grid1 = [](size_t n) { return (unsigned)((n + 255) / 256); };

    // weight prep: fp32 -> bf16 transposed
    k_transpose_bf16<<<grid1((size_t)D_H * D_H), 256, 0, stream>>>(W_x, wtx, D_H, D_H);
    k_transpose_bf16<<<grid1((size_t)D_H * D_H), 256, 0, stream>>>(Ws0, wts0, D_H, D_H);
    k_transpose_bf16<<<grid1((size_t)D_H * D_H), 256, 0, stream>>>(Wd0, wtd0, D_H, D_H);
    k_transpose_bf16<<<grid1((size_t)D_H * D_H), 256, 0, stream>>>(Wn0, wtn0, D_H, D_H);
    k_transpose_bf16<<<grid1((size_t)D_H * D_H), 256, 0, stream>>>(Ws1, wts1, D_H, D_H);
    k_transpose_bf16<<<grid1((size_t)D_H * D_H), 256, 0, stream>>>(Wd1, wtd1, D_H, D_H);
    k_transpose_bf16<<<grid1((size_t)D_H * D_H), 256, 0, stream>>>(Wn1, wtn1, D_H, D_H);
    k_transpose_bf16<<<grid1((size_t)D_H * D_OUTC), 256, 0, stream>>>(Wo, wto, D_H, D_OUTC);

    k_convert_bf16<<<grid1((size_t)N * D_H), 256, 0, stream>>>(node_feat, feat_bf, (size_t)N * D_H);

    auto gemm = [&](const __bf16* A, const __bf16* WT, const float* bias, const float* res,
                    float* Cf, __bf16* Cb, int Kout, int act) {
        int waves  = ((N + 16 * MT - 1) / (16 * MT)) * (Kout / 16);
        int blocks = (waves + 3) / 4;                 // 128 threads = 4 waves per block
        k_gemm_bf16<<<blocks, 128, 0, stream>>>(A, WT, bias, res, Cf, Cb, N, Kout, act);
    };

    // node_embed = node_feat @ W_x + b_x  (fp32 + bf16 copies)
    gemm(feat_bf, wtx, b_x, nullptr, ne, curb, D_H, 0);

    const __bf16* Wst[2] = {wts0, wts1};
    const __bf16* Wdt[2] = {wtd0, wtd1};
    const __bf16* Wnt[2] = {wtn0, wtn1};
    const float*  Vv [2] = {v0, v1};
    const float*  Bn [2] = {bn0, bn1};

    const int eblocks = (E + 7) / 8;                  // 8 waves (edges) per 256-thread block

    for (int l = 0; l < 2; ++l) {
        const float* curF = (l == 0) ? ne : curf;
        gemm(curb, Wst[l], nullptr, nullptr, sp, nullptr, D_H, 0);    // src_proj
        gemm(curb, Wdt[l], nullptr, nullptr, dp, nullptr, D_H, 0);    // dst_proj
        k_fill_u32<<<grid1((size_t)N), 256, 0, stream>>>(segmax, 0x007FFFFFu, (size_t)N); // enc(-inf)
        k_fill_u32<<<grid1((size_t)N), 256, 0, stream>>>((unsigned*)denom, 0u, (size_t)N);
        k_edge_scores<<<eblocks, 256, 0, stream>>>(sp, dp, Vv[l], edge_src, edge_dst, t, segmax, E);
        k_edge_exp<<<grid1((size_t)E), 256, 0, stream>>>(t, edge_dst, segmax, denom, E);
        // reuse dp as pool (dst_proj no longer needed)
        k_fill_u32<<<grid1((size_t)N * D_H), 256, 0, stream>>>((unsigned*)dp, 0u, (size_t)N * D_H);
        k_edge_pool<<<eblocks, 256, 0, stream>>>(t, denom, edge_src, edge_dst, curF, dp, E);
        // reuse feat_bf as bf16 pool
        k_convert_bf16<<<grid1((size_t)N * D_H), 256, 0, stream>>>(dp, feat_bf, (size_t)N * D_H);
        // cur = tanh(pool @ Wn + bn + node_embed)
        gemm(feat_bf, Wnt[l], Bn[l], ne, curf, curb, D_H, 1);
    }

    // logits = cur @ Wo + bo (reuse sp buffer), then gather selected rows
    gemm(curb, wto, bo, nullptr, sp, nullptr, D_OUTC, 0);
    k_select<<<grid1((size_t)NSEL * D_OUTC), 256, 0, stream>>>(sp, select_idx, out, NSEL);
}
